// Swin_transformerBlock_68925635167009
// MI455X (gfx1250) — matrix operand order repaired and checked by reading, not compiled, not run on hardware
//
#include <hip/hip_runtime.h>

typedef __attribute__((ext_vector_type(16))) __bf16 v16bf;
typedef __attribute__((ext_vector_type(8)))  float  v8f;

#define WSZ   7
#define SSZ   3
#define NHEAD 16
#define HD    32
#define NTOK  49
#define HH    56
#define CCH   512
#define PSTR  80   // padded LDS row stride in bf16 elems (160 B, 16B-aligned rows)

// d_ws layout: float add_table[4][16][64][64]  (bias + shift-mask + padding, 1 MB)
#define ADD_ELEMS (4 * 16 * 64 * 64)

struct __align__(16) WaveLds {
  __bf16 ps[64 * PSTR];    // softmaxed P, bf16            (10240 B)
  __bf16 vsT[32 * PSTR];   // V^T: [head_dim][token], bf16 ( 5120 B)
  int    tokbase[64];      // per-token global base offset (  256 B)
};                         // 15616 B per wave, 4 waves = 62464 B

// cyclic shift (-3,-3) + window partition + final inverse roll all reduce to
// the same coordinate map for both loads and stores:
__device__ __forceinline__ int coord_base(int b, int wh, int ww, int t) {
  int i = wh * WSZ + t / WSZ + SSZ; if (i >= HH) i -= HH;
  int j = ww * WSZ + t % WSZ + SSZ; if (j >= HH) j -= HH;
  return ((b * HH + i) * HH + j) * CCH;
}

__device__ __forceinline__ int zone1(int i) {
  return (i < HH - WSZ) ? 0 : ((i < HH - SSZ) ? 1 : 2);
}

// ---------------------------------------------------------------------------
// Setup kernel: build add[cls][head][tr][tc] = rel-pos bias + shift mask,
// with -1e30 in padding columns (tc >= 49). Only 4 mask classes exist:
// cls = (is_last_row_window ? 2 : 0) | (is_last_col_window ? 1 : 0).
// ---------------------------------------------------------------------------
__global__ void build_add_table(const float* __restrict__ bias_table,
                                float* __restrict__ addt)
{
  int idx = blockIdx.x * 256 + threadIdx.x;
  if (idx >= ADD_ELEMS) return;
  int tc   = idx & 63;
  int tr   = (idx >> 6) & 63;
  int head = (idx >> 12) & 15;
  int cls  = idx >> 16;

  float val;
  if (tc >= NTOK) {
    val = -1e30f;                                   // padding column -> weight 0
  } else {
    int trl = (tr < NTOK) ? tr : 0;                 // finite values for pad rows
    int rr = trl / WSZ, rc = trl % WSZ;
    int cr = tc  / WSZ, cc = tc  % WSZ;
    int bidx = (rr - cr + WSZ - 1) * (2 * WSZ - 1) + (rc - cc + WSZ - 1);
    float bias = bias_table[bidx * NHEAD + head];
    int whe = (cls & 2) ? (HH / WSZ - 1) : 0;
    int wwe = (cls & 1) ? (HH / WSZ - 1) : 0;
    int zr = zone1(whe * WSZ + rr) * 3 + zone1(wwe * WSZ + rc);
    int zc = zone1(whe * WSZ + cr) * 3 + zone1(wwe * WSZ + cc);
    val = bias + ((zr != zc) ? -100.f : 0.f);
  }
  addt[idx] = val;
}

// ---------------------------------------------------------------------------
// Main fused Swin window-attention kernel: one wave32 per (window, head).
// ---------------------------------------------------------------------------
__global__ __launch_bounds__(128, 1) void swin_attn_wmma(
    const float* __restrict__ q, const float* __restrict__ k,
    const float* __restrict__ v, const float* __restrict__ addt,
    float* __restrict__ out)
{
  __shared__ WaveLds lds[4];
  const int wave = threadIdx.x >> 5;
  const int lane = threadIdx.x & 31;
  const int half = lane >> 4;       // which 16-lane half of the wave
  const int lcol = lane & 15;
  WaveLds& L = lds[wave];

  const int w    = blockIdx.x * 4 + wave;   // 16384 (window, head) pairs
  const int head = w & 15;
  const int win  = w >> 4;
  const int ww   = win & 7;
  const int wh   = (win >> 3) & 7;
  const int b    = win >> 6;

  const float scale = 0.17677669529663687f;  // 1/sqrt(32)
  const int   cls  = ((wh == 7) ? 2 : 0) | ((ww == 7) ? 1 : 0);
  const float* abase = addt + ((cls * NHEAD + head) << 12);

  // ---- per-token global base offsets (shift+window+roll map), once ----
  for (int t = lane; t < 64; t += 32)
    L.tokbase[t] = (t < NTOK) ? coord_base(b, wh, ww, t) : 0;

  // ---- stage V^T (bf16) into LDS; zero the padding tokens 49..63 ----
  for (int i = lane; i < 64 * 32; i += 32) {
    int t = i >> 5, c = i & 31;
    float val = 0.f;
    if (t < NTOK) val = v[L.tokbase[t] + head * HD + c];
    L.vsT[c * PSTR + t] = (__bf16)val;
  }

  // ---- K B-fragments for QK^T: 4 column tiles of 32x16 bf16 ----
  // lane holds K row (token) n = 16*ni + lcol, k-offsets 16*half .. +15
  v16bf kf[4];
  #pragma unroll
  for (int ni = 0; ni < 4; ++ni) {
    v16bf f = {};
    int n = ni * 16 + lcol;
    if (n < NTOK) {
      const float* kp = k + L.tokbase[n] + head * HD + half * 16;
      #pragma unroll
      for (int j = 0; j < 16; ++j) f[j] = (__bf16)kp[j];
    }
    kf[ni] = f;
  }

  // ---- QK^T (+bias+mask via C operand) + softmax, one 16-row block at a time
  for (int mi = 0; mi < 4; ++mi) {
    // Q A-fragment: lane holds row t = 16*mi + lcol,
    // k elems {16*half..+15} in v[0..15]
    v16bf qa = {};
    {
      int t = mi * 16 + lcol;
      if (t < NTOK) {
        const float* qp = q + L.tokbase[t] + head * HD;
        #pragma unroll
        for (int j = 0; j < 8; ++j) qa[j]     = (__bf16)(qp[half * 16 + j] * scale);
        #pragma unroll
        for (int j = 0; j < 8; ++j) qa[8 + j] = (__bf16)(qp[half * 16 + 8 + j] * scale);
      }
    }

    v8f s[4];
    #pragma unroll
    for (int ni = 0; ni < 4; ++ni) {
      // C operand pre-loaded with bias+mask+padding in native C/D layout:
      // VGPR r, lane -> (tr = 16*mi + 8*half + r, tc = 16*ni + lcol)
      v8f cini;
      const float* ap = abase + (mi * 16 + half * 8) * 64 + ni * 16 + lcol;
      #pragma unroll
      for (int r = 0; r < 8; ++r) cini[r] = ap[r * 64];
      s[ni] = __builtin_amdgcn_wmma_f32_16x16x32_bf16(false, qa, false, kf[ni],
                                                      (short)0, cini, false, false);
    }

    // row-wise softmax; a full row lives in one 16-lane half
    #pragma unroll
    for (int r = 0; r < 8; ++r) {
      int tr = mi * 16 + half * 8 + r;

      float mx = -1e30f;
      #pragma unroll
      for (int ni = 0; ni < 4; ++ni) mx = fmaxf(mx, s[ni][r]);
      #pragma unroll
      for (int off = 8; off >= 1; off >>= 1)
        mx = fmaxf(mx, __shfl_xor(mx, off, 32));

      float sum = 0.f;
      #pragma unroll
      for (int ni = 0; ni < 4; ++ni) {
        float e = __expf(s[ni][r] - mx);
        s[ni][r] = e; sum += e;
      }
      #pragma unroll
      for (int off = 8; off >= 1; off >>= 1)
        sum += __shfl_xor(sum, off, 32);

      float inv = __builtin_amdgcn_rcpf(sum);
      #pragma unroll
      for (int ni = 0; ni < 4; ++ni)
        L.ps[tr * PSTR + ni * 16 + lcol] = (__bf16)(s[ni][r] * inv);
    }
  }

  // ---- V B-fragments for PV: B is 64x32 (tokens x head_dim), 2 K-steps x 2 N tiles
  // lane holds column n = 16*nh + lcol, k (tokens) {32*c + 16*half .. +15}
  v16bf vf[2][2];
  #pragma unroll
  for (int c = 0; c < 2; ++c)
    #pragma unroll
    for (int nh = 0; nh < 2; ++nh) {
      v16bf f;
      const __bf16* vp = &L.vsT[(nh * 16 + lcol) * PSTR + c * 32 + half * 16];
      #pragma unroll
      for (int j = 0; j < 16; ++j) f[j] = vp[j];
      vf[c][nh] = f;
    }

  // ---- out = P * V ----
  for (int mi = 0; mi < 4; ++mi) {
    v8f o0 = {}, o1 = {};
    #pragma unroll
    for (int c = 0; c < 2; ++c) {
      // P A-fragment: row M = 16*mi + lcol, k elems {32c+16*half..+15}
      v16bf pa;
      const __bf16* pp = &L.ps[(mi * 16 + lcol) * PSTR + c * 32 + half * 16];
      #pragma unroll
      for (int j = 0; j < 8; ++j) pa[j]     = pp[j];
      #pragma unroll
      for (int j = 0; j < 8; ++j) pa[8 + j] = pp[8 + j];
      o0 = __builtin_amdgcn_wmma_f32_16x16x32_bf16(false, pa, false, vf[c][0],
                                                   (short)0, o0, false, false);
      o1 = __builtin_amdgcn_wmma_f32_16x16x32_bf16(false, pa, false, vf[c][1],
                                                   (short)0, o1, false, false);
    }
    #pragma unroll
    for (int r = 0; r < 8; ++r) {
      int tr = mi * 16 + half * 8 + r;
      if (tr < NTOK) {
        float* op = out + L.tokbase[tr] + head * HD;
        op[lcol]      = o0[r];
        op[16 + lcol] = o1[r];
      }
    }
  }
}

extern "C" void kernel_launch(void* const* d_in, const int* in_sizes, int n_in,
                              void* d_out, int out_size, void* d_ws, size_t ws_size,
                              hipStream_t stream) {
  const float* q          = (const float*)d_in[0];
  const float* k          = (const float*)d_in[1];
  const float* v          = (const float*)d_in[2];
  const float* bias_table = (const float*)d_in[3];
  float* out  = (float*)d_out;
  float* addt = (float*)d_ws;   // 4*16*64*64 floats = 1 MB

  // 1) build bias+mask table (4 mask classes x 16 heads x 64x64)
  build_add_table<<<(ADD_ELEMS + 255) / 256, 256, 0, stream>>>(bias_table, addt);

  // 2) fused attention: 16 batches * 64 windows * 16 heads = 16384 waves,
  //    4 waves (128 threads) per block
  const int nblocks = (16 * 64 * 16) / 4;
  swin_attn_wmma<<<nblocks, 128, 0, stream>>>(q, k, v, addt, out);
}
